// RTDETRTransformerv2_20950850470020
// MI455X (gfx1250) — matrix operand order repaired and checked by reading, not compiled, not run on hardware
//
#include <hip/hip_runtime.h>
#include <hip/hip_bf16.h>
#include <cstdint>

typedef __attribute__((ext_vector_type(16))) _Float16 v16h;
typedef __attribute__((ext_vector_type(8)))  _Float16 h8;
typedef __attribute__((ext_vector_type(8)))  float    v8f;

#define BS 16
#define NQ 300
#define DMODEL 256
#define NH 8
#define HD 32
#define LV 8500
#define NLVL 4
#define NP 4
#define SUMP 16
#define OFFSET_SCALE 0.5f
#define NPS 0.25f

// ---------------------------------------------------------------------------
// Kernel 0: transpose + convert weights: W f32 [256 x N] row-major
//           -> Wt f16 [N x 256] (N-major, contiguous K per output column).
// One-time 128KB; makes WMMA B-fragment loads contiguous b128s.
// ---------------------------------------------------------------------------
__global__ __launch_bounds__(256)
void transpose_w_kernel(const float* __restrict__ W, _Float16* __restrict__ Wt, int N)
{
    const int n = blockIdx.x;   // output column of original W
    const int k = threadIdx.x;  // 0..255
    Wt[(size_t)n * DMODEL + k] = (_Float16)W[(size_t)k * N + n];
}

// ---------------------------------------------------------------------------
// Kernel A: v = value @ W_val + b_val, stored f16.
// A: [M,256] f32 row-major; Bt: W_val^T f16 [256 x 256] (N-major).
// Block = 64 threads = 2 waves; wave w covers cols [w*128, w*128+128).
// Wave computes a 16x128 tile: 8 K-steps x 8 N-tiles = 64 v_wmma.
// Epilogue stages the 16x256 tile in LDS for coalesced b128 stores.
// ---------------------------------------------------------------------------
__global__ __launch_bounds__(64)
void gemm_val_kernel(const float* __restrict__ A,
                     const _Float16* __restrict__ Bt,
                     const float* __restrict__ bias,
                     _Float16* __restrict__ Vout)
{
    constexpr int LSTR = 272;                 // halfs; 544B row stride (16B mult)
    __shared__ _Float16 stile[16 * LSTR];

    const int lane = threadIdx.x & 31;
    const int wave = threadIdx.x >> 5;
    const int m0   = blockIdx.x * 16;
    const int n0   = wave * 128;
    const int row  = lane & 15;
    const int hi   = lane >> 4;

    v8f acc[8];
#pragma unroll
    for (int i = 0; i < 8; ++i) acc[i] = (v8f){0.f,0.f,0.f,0.f,0.f,0.f,0.f,0.f};

    for (int kk = 0; kk < 8; ++kk) {
        const int k0 = kk * 32;
        // A fragment (16x32 f16): two 8-float chunks per lane, cvt on the fly
        v16h afrag;
        {
            const float* ap = A + (size_t)(m0 + row) * DMODEL + k0 + hi * 8;
            const float4 a0 = *(const float4*)(ap);
            const float4 a1 = *(const float4*)(ap + 4);
            const float4 a2 = *(const float4*)(ap + 16);
            const float4 a3 = *(const float4*)(ap + 20);
            afrag[0]  = (_Float16)a0.x; afrag[1]  = (_Float16)a0.y;
            afrag[2]  = (_Float16)a0.z; afrag[3]  = (_Float16)a0.w;
            afrag[4]  = (_Float16)a1.x; afrag[5]  = (_Float16)a1.y;
            afrag[6]  = (_Float16)a1.z; afrag[7]  = (_Float16)a1.w;
            afrag[8]  = (_Float16)a2.x; afrag[9]  = (_Float16)a2.y;
            afrag[10] = (_Float16)a2.z; afrag[11] = (_Float16)a2.w;
            afrag[12] = (_Float16)a3.x; afrag[13] = (_Float16)a3.y;
            afrag[14] = (_Float16)a3.z; afrag[15] = (_Float16)a3.w;
        }
#pragma unroll
        for (int nt = 0; nt < 8; ++nt) {
            const int ncol = n0 + nt * 16 + (lane & 15);
            // B fragment (32x16 f16): lane = column, K chunks +8h..+8h+7 and +16+8h..+16+8h+7
            const h8* bp = (const h8*)(Bt + (size_t)ncol * DMODEL + k0 + hi * 8);
            const h8 b0 = bp[0];
            const h8 b1 = bp[2];
            const v16h bfrag = __builtin_shufflevector(
                b0, b1, 0,1,2,3,4,5,6,7,8,9,10,11,12,13,14,15);
            acc[nt] = __builtin_amdgcn_wmma_f32_16x16x32_f16(
                false, afrag, false, bfrag, (short)0, acc[nt], false, false);
        }
    }

    // Epilogue: bias + stage in LDS, then coalesced b128 stores
#pragma unroll
    for (int nt = 0; nt < 8; ++nt) {
        const int ncol = n0 + nt * 16 + (lane & 15);
        const float bv = bias[ncol];
#pragma unroll
        for (int r = 0; r < 8; ++r)
            stile[(r + 8 * hi) * LSTR + ncol] = (_Float16)(acc[nt][r] + bv);
    }
    __syncthreads();
    {
        const int trow = threadIdx.x >> 2;          // 0..15
        const int colq = (threadIdx.x & 3) * 64;    // 0,64,128,192
        _Float16* gp = Vout + (size_t)(m0 + trow) * DMODEL + colq;
        const _Float16* lp = stile + trow * LSTR + colq;
#pragma unroll
        for (int i = 0; i < 8; ++i)
            *(h8*)(gp + i * 8) = *(const h8*)(lp + i * 8);
    }
}

// ---------------------------------------------------------------------------
// Kernel B: sampling offsets + attention logits + softmax per (b,q) row.
// ---------------------------------------------------------------------------
__global__ __launch_bounds__(64)
void offattn_kernel(const float* __restrict__ query,
                    const float* __restrict__ W_off,
                    const float* __restrict__ b_off,
                    const float* __restrict__ W_attn,
                    const float* __restrict__ b_attn,
                    float* __restrict__ off_out,
                    float* __restrict__ attn_out)
{
    __shared__ float qrow[DMODEL];
    __shared__ float logits[SUMP];
    const int bq = blockIdx.x;
    const float* q = query + (size_t)bq * DMODEL;
    for (int i = threadIdx.x; i < DMODEL; i += 64) qrow[i] = q[i];
    __syncthreads();

    const int t = threadIdx.x;
    if (t < 32) {
        float s = b_off[t];
        for (int k = 0; k < DMODEL; ++k) s += qrow[k] * W_off[k * (SUMP * 2) + t];
        off_out[(size_t)bq * 32 + t] = s;
    } else if (t < 48) {
        const int n = t - 32;
        float s = b_attn[n];
        for (int k = 0; k < DMODEL; ++k) s += qrow[k] * W_attn[k * SUMP + n];
        logits[n] = s;
    }
    __syncthreads();
    if (t >= 32 && t < 48) {
        const int n = t - 32;
        float mx = logits[0];
        for (int i = 1; i < SUMP; ++i) mx = fmaxf(mx, logits[i]);
        float sum = 0.f;
        for (int i = 0; i < SUMP; ++i) sum += expf(logits[i] - mx);
        attn_out[(size_t)bq * SUMP + n] = expf(logits[n] - mx) / sum;
    }
}

// ---------------------------------------------------------------------------
// Kernel C: bilinear sampling + attention-weighted sum.
// Block = 256 threads = one (b,q); thread = (h, hd): 32 consecutive lanes
// share a head -> each corner gather is 32 consecutive f16 (64B).
// ---------------------------------------------------------------------------
__global__ __launch_bounds__(256)
void sample_kernel(const float* __restrict__ refpts,
                   const float* __restrict__ off_ws,
                   const float* __restrict__ attn_ws,
                   const _Float16* __restrict__ vproj,
                   _Float16* __restrict__ hout)
{
    __shared__ float sOff[SUMP * 2];
    __shared__ float sAttn[SUMP];
    __shared__ float sRef[4];
    const int bq = blockIdx.x;
    const int b  = bq / NQ;
    const int t  = threadIdx.x;
    const int h  = t >> 5;
    const int hd = t & 31;

    if (t < 32) sOff[t]  = off_ws[(size_t)bq * 32 + t];
    if (t < 16) sAttn[t] = attn_ws[(size_t)bq * SUMP + t];
    if (t < 4)  sRef[t]  = refpts[(size_t)bq * 4 + t];
    __syncthreads();

    const float rx = sRef[0], ry = sRef[1], rw = sRef[2], rh = sRef[3];
    const int   HS[NLVL] = {80, 40, 20, 10};
    const int   WS[NLVL] = {80, 40, 20, 10};
    const int   SV[NLVL] = {0, 6400, 8000, 8400};

    const size_t vbase = (size_t)b * LV * DMODEL;
    const int chan = h * HD + hd;
    float acc = 0.f;

#pragma unroll
    for (int lvl = 0; lvl < NLVL; ++lvl) {
        const int lh = HS[lvl], lw = WS[lvl];
        const size_t lvbase = vbase + (size_t)SV[lvl] * DMODEL;
#pragma unroll
        for (int p = 0; p < NP; ++p) {
            const int sp = lvl * NP + p;
            const float aw = sAttn[sp];
            const float cx = rx + sOff[sp * 2 + 0] * (NPS * OFFSET_SCALE) * rw;
            const float cy = ry + sOff[sp * 2 + 1] * (NPS * OFFSET_SCALE) * rh;
            const float x = cx * (float)lw - 0.5f;
            const float y = cy * (float)lh - 0.5f;
            const float x0f = floorf(x), y0f = floorf(y);
            const float lx = x - x0f, ly = y - y0f;
            const int x0 = (int)x0f, y0 = (int)y0f;
#pragma unroll
            for (int dy = 0; dy < 2; ++dy) {
                const int yi = y0 + dy;
                const float wy = dy ? ly : (1.f - ly);
                const int yc = yi < 0 ? 0 : (yi > lh - 1 ? lh - 1 : yi);
                const bool vy = (yi >= 0) && (yi < lh);
#pragma unroll
                for (int dx = 0; dx < 2; ++dx) {
                    const int xi = x0 + dx;
                    const float wx = dx ? lx : (1.f - lx);
                    const int xc = xi < 0 ? 0 : (xi > lw - 1 ? lw - 1 : xi);
                    const bool ok = vy && (xi >= 0) && (xi < lw);
                    const float wgt = ok ? (wy * wx) : 0.f;
                    const float val =
                        (float)vproj[lvbase + (size_t)(yc * lw + xc) * DMODEL + chan];
                    acc += aw * wgt * val;
                }
            }
        }
    }
    hout[(size_t)bq * DMODEL + t] = (_Float16)acc;
}

// ---------------------------------------------------------------------------
// Kernel D: out = h @ W_out + b_out. A f16 [4800,256], Bt = W_out^T f16,
// output f32 with LDS-staged coalesced stores.
// ---------------------------------------------------------------------------
__global__ __launch_bounds__(64)
void gemm_out_kernel(const _Float16* __restrict__ A,
                     const _Float16* __restrict__ Bt,
                     const float* __restrict__ bias,
                     float* __restrict__ Out)
{
    constexpr int LSTR = 264;                 // floats; 1056B row stride (16B mult)
    __shared__ float stile[16 * LSTR];

    const int lane = threadIdx.x & 31;
    const int wave = threadIdx.x >> 5;
    const int m0   = blockIdx.x * 16;
    const int n0   = wave * 128;
    const int row  = lane & 15;
    const int hi   = lane >> 4;

    v8f acc[8];
#pragma unroll
    for (int i = 0; i < 8; ++i) acc[i] = (v8f){0.f,0.f,0.f,0.f,0.f,0.f,0.f,0.f};

    for (int kk = 0; kk < 8; ++kk) {
        const int k0 = kk * 32;
        v16h afrag;
        {
            const h8* ap = (const h8*)(A + (size_t)(m0 + row) * DMODEL + k0 + hi * 8);
            const h8 a0 = ap[0];   // K chunk +0..7
            const h8 a1 = ap[2];   // K chunk +16..23
            afrag = __builtin_shufflevector(
                a0, a1, 0,1,2,3,4,5,6,7,8,9,10,11,12,13,14,15);
        }
#pragma unroll
        for (int nt = 0; nt < 8; ++nt) {
            const int ncol = n0 + nt * 16 + (lane & 15);
            const h8* bp = (const h8*)(Bt + (size_t)ncol * DMODEL + k0 + hi * 8);
            const h8 b0 = bp[0];
            const h8 b1 = bp[2];
            const v16h bfrag = __builtin_shufflevector(
                b0, b1, 0,1,2,3,4,5,6,7,8,9,10,11,12,13,14,15);
            acc[nt] = __builtin_amdgcn_wmma_f32_16x16x32_f16(
                false, afrag, false, bfrag, (short)0, acc[nt], false, false);
        }
    }

#pragma unroll
    for (int nt = 0; nt < 8; ++nt) {
        const int ncol = n0 + nt * 16 + (lane & 15);
        const float bv = bias[ncol];
#pragma unroll
        for (int r = 0; r < 8; ++r)
            stile[(r + 8 * hi) * LSTR + ncol] = acc[nt][r] + bv;
    }
    __syncthreads();
    {
        const int trow = threadIdx.x >> 2;
        const int colq = (threadIdx.x & 3) * 64;
        float* gp = Out + (size_t)(m0 + trow) * DMODEL + colq;
        const float* lp = stile + trow * LSTR + colq;
#pragma unroll
        for (int i = 0; i < 16; ++i)
            *(float4*)(gp + i * 4) = *(const float4*)(lp + i * 4);
    }
}

extern "C" void kernel_launch(void* const* d_in, const int* in_sizes, int n_in,
                              void* d_out, int out_size, void* d_ws, size_t ws_size,
                              hipStream_t stream)
{
    (void)in_sizes; (void)n_in; (void)out_size; (void)ws_size;

    const float* query   = (const float*)d_in[0];   // [16,300,256]
    const float* refpts  = (const float*)d_in[1];   // [16,300,4]
    const float* value   = (const float*)d_in[2];   // [16,8500,256]
    const float* W_val   = (const float*)d_in[3];   // [256,256]
    const float* b_val   = (const float*)d_in[4];   // [256]
    const float* W_off   = (const float*)d_in[5];   // [256,32]
    const float* b_off   = (const float*)d_in[6];   // [32]
    const float* W_attn  = (const float*)d_in[7];   // [256,16]
    const float* b_attn  = (const float*)d_in[8];   // [16]
    const float* W_out   = (const float*)d_in[9];   // [256,256]
    const float* b_out   = (const float*)d_in[10];  // [256]
    float* out = (float*)d_out;                     // [16,300,256]

    // Workspace carve-up
    char* ws = (char*)d_ws;
    _Float16* vproj = (_Float16*)ws;                         // 69.6 MB
    ws += (size_t)BS * LV * DMODEL * sizeof(_Float16);
    float* off_ws = (float*)ws;
    ws += (size_t)BS * NQ * SUMP * 2 * sizeof(float);
    float* attn_ws = (float*)ws;
    ws += (size_t)BS * NQ * SUMP * sizeof(float);
    _Float16* hout = (_Float16*)ws;
    ws += (size_t)BS * NQ * DMODEL * sizeof(_Float16);
    _Float16* bt_val = (_Float16*)ws;                        // W_val^T f16
    ws += (size_t)DMODEL * DMODEL * sizeof(_Float16);
    _Float16* bt_out = (_Float16*)ws;                        // W_out^T f16

    // 0) weight transpose+convert (one-time, tiny)
    transpose_w_kernel<<<DMODEL, DMODEL, 0, stream>>>(W_val, bt_val, DMODEL);
    transpose_w_kernel<<<DMODEL, DMODEL, 0, stream>>>(W_out, bt_out, DMODEL);

    // 1) value projection: M = 136000 rows -> 8500 M-tiles of 16
    gemm_val_kernel<<<(BS * LV) / 16, 64, 0, stream>>>(value, bt_val, b_val, vproj);

    // 2) offsets + attention softmax per (b,q)
    offattn_kernel<<<BS * NQ, 64, 0, stream>>>(query, W_off, b_off, W_attn, b_attn,
                                               off_ws, attn_ws);

    // 3) bilinear sampling + weighted sum per (b,q)
    sample_kernel<<<BS * NQ, 256, 0, stream>>>(refpts, off_ws, attn_ws, vproj, hout);

    // 4) output projection: M = 4800 rows -> 300 M-tiles of 16
    gemm_out_kernel<<<(BS * NQ) / 16, 64, 0, stream>>>(hout, bt_out, b_out, out);
}
